// PDeep2_26525718020276
// MI455X (gfx1250) — matrix operand order repaired and checked by reading, not compiled, not run on hardware
//
#include <hip/hip_runtime.h>
#include <cstdint>
#include <cstddef>

// ---------------------------------------------------------------------------
// Types for CDNA5 WMMA (wave32, v_wmma_f32_16x16x32_bf16)
// ---------------------------------------------------------------------------
typedef __bf16 bf16;
typedef __attribute__((ext_vector_type(8)))  float        v8f;
typedef __attribute__((ext_vector_type(16))) bf16         v16bf;
typedef __attribute__((ext_vector_type(4)))  unsigned int u32x4;

union Frag { v16bf v; u32x4 u[2]; };

#define B_SZ     1024
#define T_STEPS  31
#define HID      256
#define HSTRIDE  264   // 256 + 8 bf16 pad -> rotates LDS banks across rows

__device__ __forceinline__ float sigf(float x) { return 1.f / (1.f + __expf(-x)); }

// ---------------------------------------------------------------------------
// Pack f32 weight matrix W (G x Kreal, row-major; gates x in-dim) into the
// WMMA B-fragment layout: P[ks][ntile][lane][e] (bf16), where
//   N = ntile*16 + (lane&15),  K = ks*32 + (e>>3)*16 + (lane>>4)*8 + (e&7)
// so each lane's 16 elements are contiguous (2x global_load_b128 per frag).
// Zero-pads K beyond Kreal.
// ---------------------------------------------------------------------------
__global__ void pack_b_kernel(const float* __restrict__ W, bf16* __restrict__ P,
                              int G, int Kreal, int ksteps) {
  int idx = blockIdx.x * blockDim.x + threadIdx.x;
  int total = ksteps * 64 * 32 * 16;
  if (idx >= total) return;
  int e    = idx & 15;
  int lane = (idx >> 4) & 31;
  int nt   = (idx >> 9) & 63;
  int ks   = idx >> 15;
  int n = nt * 16 + (lane & 15);
  int k = ks * 32 + (e >> 3) * 16 + (lane >> 4) * 8 + (e & 7);
  float v = (n < G && k < Kreal) ? W[(size_t)n * Kreal + k] : 0.f;
  P[idx] = (bf16)v;
}

__global__ void bias_kernel(const float* __restrict__ bih,
                            const float* __restrict__ bhh,
                            float* __restrict__ bias, int n) {
  int i = blockIdx.x * blockDim.x + threadIdx.x;
  if (i < n) bias[i] = bih[i] + bhh[i];
}

// ---------------------------------------------------------------------------
// Per-batch extras: charge = argmax(charges[b,0:6]);
// proj_inst = [inst(8), nce] @ w_proj.T where inst = ones iff b < 5.
// ---------------------------------------------------------------------------
__global__ void extras_kernel(const float* __restrict__ charges,
                              const float* __restrict__ nce,
                              const float* __restrict__ wproj,
                              float* __restrict__ extras) {
  int b = blockIdx.x * blockDim.x + threadIdx.x;
  if (b >= B_SZ) return;
  const float* ch = charges + b * 6;
  int am = 0; float mv = ch[0];
  for (int i = 1; i < 6; ++i) { if (ch[i] > mv) { mv = ch[i]; am = i; } }
  extras[b * 4 + 0] = (float)am;
  float n = nce[b];
  for (int r = 0; r < 3; ++r) {
    float a = 0.f;
    if (b < 5) for (int c = 0; c < 8; ++c) a += wproj[r * 9 + c];
    a += n * wproj[r * 9 + 8];
    extras[b * 4 + 1 + r] = a;
  }
}

// ---------------------------------------------------------------------------
// Build layer-1 input features: X1 (B, T, 96) bf16 (94 real dims, 2 pad).
// [0:22) prev one-hot, [22:44) next one-hot, [44:66) prefix counts,
// [66:88) suffix (zeros), 88 flag_first, 89 flag_last, 90 charge, 91:94 proj.
// ---------------------------------------------------------------------------
__global__ void featurize_kernel(const int* __restrict__ seq,
                                 const int* __restrict__ lengths,
                                 const float* __restrict__ extras,
                                 bf16* __restrict__ X1) {
  int id = blockIdx.x * blockDim.x + threadIdx.x;
  if (id >= B_SZ * T_STEPS) return;
  int b = id / T_STEPS, t = id % T_STEPS;
  bf16* x = X1 + (size_t)id * 96;
  const int* s = seq + b * 32;
  int st = s[t], sn = s[t + 1];
  for (int d = 0; d < 22; ++d) {
    x[d]      = (bf16)(float)(st == d);
    x[22 + d] = (bf16)(float)(sn == d);
    int cnt = 0;
    for (int j = 0; j < t; ++j) cnt += (s[j] == d);
    x[44 + d] = (bf16)(float)cnt;
    x[66 + d] = (bf16)0.f;
  }
  x[88] = (bf16)(t == 0 ? 1.f : 0.f);
  x[89] = (bf16)((lengths[b] - 2 == t) ? 1.f : 0.f);
  x[90] = (bf16)extras[b * 4 + 0];
  x[91] = (bf16)extras[b * 4 + 1];
  x[92] = (bf16)extras[b * 4 + 2];
  x[93] = (bf16)extras[b * 4 + 3];
  x[94] = (bf16)0.f;
  x[95] = (bf16)0.f;
}

// Fill columns [512:544) of X2 and X3: charge, proj(3), then zero pad.
__global__ void fill_extras_kernel(const float* __restrict__ extras,
                                   bf16* __restrict__ X2, bf16* __restrict__ X3) {
  int id = blockIdx.x * blockDim.x + threadIdx.x;
  if (id >= B_SZ * T_STEPS) return;
  int b = id / T_STEPS;
  bf16* p2 = X2 + (size_t)id * 544 + 512;
  bf16* p3 = X3 + (size_t)id * 544 + 512;
  for (int k = 0; k < 4; ++k) {
    bf16 v = (bf16)extras[b * 4 + k];
    p2[k] = v; p3[k] = v;
  }
  for (int k = 4; k < 32; ++k) { p2[k] = (bf16)0.f; p3[k] = (bf16)0.f; }
}

// ---------------------------------------------------------------------------
// Fused LSTM direction scan (layers 1 & 2): per workgroup, 16 batch rows;
// per step: gates(16x1024) = Xt(16xKp) @ WihT + h(16x256) @ WhhT + bias,
// all via v_wmma_f32_16x16x32_bf16, then the pointwise cell update.
// Wave w owns hidden cols [w*32, w*32+32): for each gate g in {i,f,g,o} the
// two 16-wide N-tiles g*16 + w*2 + {0,1}. c stays in VGPRs across time;
// h round-trips through LDS (bank-conflict-padded) as next step's A-matrix.
// ---------------------------------------------------------------------------
__global__ __launch_bounds__(256)
void lstm_scan_kernel(const bf16* __restrict__ Xin, int Kp, int ksin,
                      const bf16* __restrict__ Wp,    // packed input weights
                      const bf16* __restrict__ Up,    // packed recurrent weights (8 ksteps)
                      const float* __restrict__ bias, // bih+bhh, 1024
                      bf16* __restrict__ Xout, int KpOut, int dirOff,
                      int reverse) {
  __shared__ __align__(16) bf16 hbuf[16 * HSTRIDE];
  const int lane = threadIdx.x & 31;
  const int w    = threadIdx.x >> 5;   // wave 0..7
  const int p    = lane >> 4;          // half-wave
  const int nl   = lane & 15;
  const int b0   = blockIdx.x * 16;

  float creg[2][8];
#pragma unroll
  for (int s = 0; s < 2; ++s)
#pragma unroll
    for (int r = 0; r < 8; ++r) creg[s][r] = 0.f;

  for (int stp = 0; stp < T_STEPS; ++stp) {
    const int tt = reverse ? (T_STEPS - 1 - stp) : stp;

    v8f acc[4][2];
#pragma unroll
    for (int g = 0; g < 4; ++g)
#pragma unroll
      for (int s = 0; s < 2; ++s) {
        float bv = bias[g * 256 + w * 32 + s * 16 + nl];
#pragma unroll
        for (int r = 0; r < 8; ++r) acc[g][s][r] = bv;
      }

    // ---- input projection: accumulate over Kp (dynamic trip count) ----
    for (int ks = 0; ks < ksin; ++ks) {
      Frag a;
      const bf16* ap = Xin + ((size_t)(b0 + nl) * T_STEPS + tt) * Kp + ks * 32 + p * 8;
      a.u[0] = *reinterpret_cast<const u32x4*>(ap);
      a.u[1] = *reinterpret_cast<const u32x4*>(ap + 16);
#pragma unroll
      for (int g = 0; g < 4; ++g)
#pragma unroll
        for (int s = 0; s < 2; ++s) {
          int nt = g * 16 + w * 2 + s;
          Frag bm;
          const bf16* bp = Wp + (((size_t)ks * 64 + nt) * 32 + lane) * 16;
          bm.u[0] = *reinterpret_cast<const u32x4*>(bp);
          bm.u[1] = *reinterpret_cast<const u32x4*>(bp + 8);
          acc[g][s] = __builtin_amdgcn_wmma_f32_16x16x32_bf16(
              false, a.v, false, bm.v, (short)0, acc[g][s], false, false);
        }
    }

    // ---- recurrent projection over K=256 (h == 0 at first step) ----
    if (stp > 0) {
      for (int ks = 0; ks < 8; ++ks) {
        Frag a;
        const bf16* ap = &hbuf[nl * HSTRIDE + ks * 32 + p * 8];
        a.u[0] = *reinterpret_cast<const u32x4*>(ap);
        a.u[1] = *reinterpret_cast<const u32x4*>(ap + 16);
#pragma unroll
        for (int g = 0; g < 4; ++g)
#pragma unroll
          for (int s = 0; s < 2; ++s) {
            int nt = g * 16 + w * 2 + s;
            Frag bm;
            const bf16* bp = Up + (((size_t)ks * 64 + nt) * 32 + lane) * 16;
            bm.u[0] = *reinterpret_cast<const u32x4*>(bp);
            bm.u[1] = *reinterpret_cast<const u32x4*>(bp + 8);
            acc[g][s] = __builtin_amdgcn_wmma_f32_16x16x32_bf16(
                false, a.v, false, bm.v, (short)0, acc[g][s], false, false);
          }
      }
    }

    __syncthreads();  // all waves done reading old h from LDS

    // ---- cell update: i,f,g,o resident in this wave's accumulators ----
#pragma unroll
    for (int s = 0; s < 2; ++s)
#pragma unroll
      for (int r = 0; r < 8; ++r) {
        float gi = acc[0][s][r];
        float gf = acc[1][s][r];
        float gg = acc[2][s][r];
        float go = acc[3][s][r];
        float cn = sigf(gf) * creg[s][r] + sigf(gi) * tanhf(gg);
        creg[s][r] = cn;
        float hv = sigf(go) * tanhf(cn);
        int m    = r + p * 8;                 // batch row within tile
        int ncol = w * 32 + s * 16 + nl;      // hidden channel
        bf16 hb = (bf16)hv;
        hbuf[m * HSTRIDE + ncol] = hb;
        Xout[((size_t)(b0 + m) * T_STEPS + tt) * KpOut + dirOff + ncol] = hb;
      }

    __syncthreads();  // new h visible before next step's reads
  }
}

// ---------------------------------------------------------------------------
// Layer 3: hidden=6 (24 gates) — register-resident scalar scan, one thread
// per batch row, both directions; writes fwd then adds bwd into d_out (f32).
// ---------------------------------------------------------------------------
__global__ __launch_bounds__(256)
void lstm_small_kernel(const bf16* __restrict__ X3,
                       const float* __restrict__ WihF, const float* __restrict__ WhhF,
                       const float* __restrict__ bihF, const float* __restrict__ bhhF,
                       const float* __restrict__ WihB, const float* __restrict__ WhhB,
                       const float* __restrict__ bihB, const float* __restrict__ bhhB,
                       float* __restrict__ out) {
  int b = blockIdx.x * blockDim.x + threadIdx.x;
  if (b >= B_SZ) return;
  for (int dir = 0; dir < 2; ++dir) {
    const float* Wih = dir ? WihB : WihF;
    const float* Whh = dir ? WhhB : WhhF;
    const float* bih = dir ? bihB : bihF;
    const float* bhh = dir ? bhhB : bhhF;
    float h[6], c[6];
#pragma unroll
    for (int j = 0; j < 6; ++j) { h[j] = 0.f; c[j] = 0.f; }
    for (int stp = 0; stp < T_STEPS; ++stp) {
      int tt = dir ? (T_STEPS - 1 - stp) : stp;
      const bf16* x = X3 + ((size_t)b * T_STEPS + tt) * 544;
      float gacc[24];
#pragma unroll
      for (int g = 0; g < 24; ++g) gacc[g] = bih[g] + bhh[g];
      for (int k = 0; k < 516; ++k) {
        float xv = (float)x[k];
#pragma unroll
        for (int g = 0; g < 24; ++g) gacc[g] += xv * Wih[g * 516 + k];
      }
#pragma unroll
      for (int k2 = 0; k2 < 6; ++k2) {
        float hv = h[k2];
#pragma unroll
        for (int g = 0; g < 24; ++g) gacc[g] += hv * Whh[g * 6 + k2];
      }
      float hn[6];
#pragma unroll
      for (int j = 0; j < 6; ++j) {
        float cn = sigf(gacc[6 + j]) * c[j] + sigf(gacc[j]) * tanhf(gacc[12 + j]);
        c[j] = cn;
        hn[j] = sigf(gacc[18 + j]) * tanhf(cn);
      }
#pragma unroll
      for (int j = 0; j < 6; ++j) h[j] = hn[j];
      float* o = out + (size_t)b * (T_STEPS * 6) + tt * 6;
#pragma unroll
      for (int j = 0; j < 6; ++j) {
        if (dir == 0) o[j] = hn[j]; else o[j] += hn[j];
      }
    }
  }
}

// ---------------------------------------------------------------------------
// Host: d_in leaf order (pytree-flattened):
// 0 sequences(i32 B*32) 1 lengths(i32 B) 2 nce(f32 B) 3 charges(f32 B*6)
// 4 w_proj(f32 27) 5..12 params1 fwd/bwd {Wih,Whh,bih,bhh}
// 13..20 params2, 21..28 params3
// ---------------------------------------------------------------------------
extern "C" void kernel_launch(void* const* d_in, const int* in_sizes, int n_in,
                              void* d_out, int out_size, void* d_ws, size_t ws_size,
                              hipStream_t stream) {
  (void)in_sizes; (void)n_in; (void)out_size; (void)ws_size;
  const int*   seq     = (const int*)d_in[0];
  const int*   lengths = (const int*)d_in[1];
  const float* nce     = (const float*)d_in[2];
  const float* charges = (const float*)d_in[3];
  const float* wproj   = (const float*)d_in[4];
  const float* Wih1[2] = { (const float*)d_in[5],  (const float*)d_in[9]  };
  const float* Whh1[2] = { (const float*)d_in[6],  (const float*)d_in[10] };
  const float* bih1[2] = { (const float*)d_in[7],  (const float*)d_in[11] };
  const float* bhh1[2] = { (const float*)d_in[8],  (const float*)d_in[12] };
  const float* Wih2[2] = { (const float*)d_in[13], (const float*)d_in[17] };
  const float* Whh2[2] = { (const float*)d_in[14], (const float*)d_in[18] };
  const float* bih2[2] = { (const float*)d_in[15], (const float*)d_in[19] };
  const float* bhh2[2] = { (const float*)d_in[16], (const float*)d_in[20] };
  const float* Wih3[2] = { (const float*)d_in[21], (const float*)d_in[25] };
  const float* Whh3[2] = { (const float*)d_in[22], (const float*)d_in[26] };
  const float* bih3[2] = { (const float*)d_in[23], (const float*)d_in[27] };
  const float* bhh3[2] = { (const float*)d_in[24], (const float*)d_in[28] };
  float* out = (float*)d_out;

  char* ws = (char*)d_ws;
  size_t off = 0;
  auto bump = [&](size_t n) { size_t o = off; off += (n + 255) & ~(size_t)255; return o; };

  bf16*  X1     = (bf16*)(ws + bump((size_t)B_SZ * T_STEPS * 96 * 2));
  bf16*  X2     = (bf16*)(ws + bump((size_t)B_SZ * T_STEPS * 544 * 2));
  bf16*  X3     = (bf16*)(ws + bump((size_t)B_SZ * T_STEPS * 544 * 2));
  float* extras = (float*)(ws + bump((size_t)B_SZ * 4 * sizeof(float)));
  bf16 *Wih1p[2], *Whh1p[2], *Wih2p[2], *Whh2p[2];
  float *b1[2], *b2[2];
  for (int d = 0; d < 2; ++d) {
    Wih1p[d] = (bf16*)(ws + bump(3ull  * 32768 * 2));
    Whh1p[d] = (bf16*)(ws + bump(8ull  * 32768 * 2));
    Wih2p[d] = (bf16*)(ws + bump(17ull * 32768 * 2));
    Whh2p[d] = (bf16*)(ws + bump(8ull  * 32768 * 2));
    b1[d]    = (float*)(ws + bump(1024 * sizeof(float)));
    b2[d]    = (float*)(ws + bump(1024 * sizeof(float)));
  }

  // ---- weight repack + bias combine ----
  for (int d = 0; d < 2; ++d) {
    pack_b_kernel<<<(3  * 32768) / 256, 256, 0, stream>>>(Wih1[d], Wih1p[d], 1024, 94,  3);
    pack_b_kernel<<<(8  * 32768) / 256, 256, 0, stream>>>(Whh1[d], Whh1p[d], 1024, 256, 8);
    pack_b_kernel<<<(17 * 32768) / 256, 256, 0, stream>>>(Wih2[d], Wih2p[d], 1024, 516, 17);
    pack_b_kernel<<<(8  * 32768) / 256, 256, 0, stream>>>(Whh2[d], Whh2p[d], 1024, 256, 8);
    bias_kernel<<<4, 256, 0, stream>>>(bih1[d], bhh1[d], b1[d], 1024);
    bias_kernel<<<4, 256, 0, stream>>>(bih2[d], bhh2[d], b2[d], 1024);
  }

  // ---- features ----
  extras_kernel<<<4, 256, 0, stream>>>(charges, nce, wproj, extras);
  featurize_kernel<<<(B_SZ * T_STEPS + 255) / 256, 256, 0, stream>>>(seq, lengths, extras, X1);
  fill_extras_kernel<<<(B_SZ * T_STEPS + 255) / 256, 256, 0, stream>>>(extras, X2, X3);

  // ---- layer 1 (94 -> 256, bidirectional) ----
  lstm_scan_kernel<<<64, 256, 0, stream>>>(X1, 96, 3,  Wih1p[0], Whh1p[0], b1[0], X2, 544, 0,   0);
  lstm_scan_kernel<<<64, 256, 0, stream>>>(X1, 96, 3,  Wih1p[1], Whh1p[1], b1[1], X2, 544, 256, 1);
  // ---- layer 2 (516 -> 256, bidirectional) ----
  lstm_scan_kernel<<<64, 256, 0, stream>>>(X2, 544, 17, Wih2p[0], Whh2p[0], b2[0], X3, 544, 0,   0);
  lstm_scan_kernel<<<64, 256, 0, stream>>>(X2, 544, 17, Wih2p[1], Whh2p[1], b2[1], X3, 544, 256, 1);
  // ---- layer 3 (516 -> 6, bidirectional, scalar) + fwd+bwd sum ----
  lstm_small_kernel<<<4, 256, 0, stream>>>(X3, Wih3[0], Whh3[0], bih3[0], bhh3[0],
                                           Wih3[1], Whh3[1], bih3[1], bhh3[1], out);
}
